// Transformer_28458453303814
// MI455X (gfx1250) — matrix operand order repaired and checked by reading, not compiled, not run on hardware
//
#include <hip/hip_runtime.h>
#include <math.h>

// ---------------------------------------------------------------------------
// Types for CDNA5 WMMA (wave32): v_wmma_f32_16x16x32_bf16
// ---------------------------------------------------------------------------
typedef __bf16 bf16;
typedef __attribute__((ext_vector_type(8)))  __bf16 v8bf;
typedef __attribute__((ext_vector_type(16))) __bf16 v16bf;
typedef __attribute__((ext_vector_type(8)))  float  v8f;
typedef __attribute__((ext_vector_type(4)))  unsigned int u32x4;
typedef __attribute__((ext_vector_type(8)))  int i32x8;
typedef __attribute__((ext_vector_type(4)))  int i32x4;

#define DEV __device__ __forceinline__

DEV v16bf cat8(v8bf lo, v8bf hi) {
    return __builtin_shufflevector(lo, hi, 0,1,2,3,4,5,6,7,8,9,10,11,12,13,14,15);
}

// A fragment (16x32 bf16, row-major source): element e -> K = 8*half+e (e<8),
// 16+8*half+(e-8) (e>=8). Two contiguous 16B loads.
DEV v16bf load_afrag(const bf16* __restrict__ p, int half) {
    v8bf lo = *reinterpret_cast<const v8bf*>(p + 8 * half);
    v8bf hi = *reinterpret_cast<const v8bf*>(p + 16 + 8 * half);
    return cat8(lo, hi);
}

// B fragment (32x16 bf16, K-contiguous source): element e -> K = 8*half+e (e<8),
// 16+8*half+(e-8) (e>=8).
DEV v16bf load_bfrag(const bf16* __restrict__ p, int half) {
    v8bf lo = *reinterpret_cast<const v8bf*>(p + 8 * half);
    v8bf hi = *reinterpret_cast<const v8bf*>(p + 16 + 8 * half);
    return cat8(lo, hi);
}

DEV v8f wmma_bf16(v16bf a, v16bf b, v8f c) {
    return __builtin_amdgcn_wmma_f32_16x16x32_bf16(false, a, false, b, (short)0, c, false, false);
}

DEV float halfmax(float v) {
    v = fmaxf(v, __shfl_xor(v, 1, 32));
    v = fmaxf(v, __shfl_xor(v, 2, 32));
    v = fmaxf(v, __shfl_xor(v, 4, 32));
    v = fmaxf(v, __shfl_xor(v, 8, 32));
    return v;
}
DEV float halfsum(float v) {
    v += __shfl_xor(v, 1, 32);
    v += __shfl_xor(v, 2, 32);
    v += __shfl_xor(v, 4, 32);
    v += __shfl_xor(v, 8, 32);
    return v;
}

// ---------------------------------------------------------------------------
// Tensor Data Mover: 2D tile (data_size = 2 bytes) global -> LDS.
// D# packing per CDNA5 ISA.
//   tile_d0 = contiguous elements per row, tile_d1 = rows,
//   stride0 = row stride in elements.
// ---------------------------------------------------------------------------
DEV void tdm_load_2d_bf16(unsigned int lds_off, unsigned long long gaddr,
                          unsigned int tile_d0, unsigned int tile_d1,
                          unsigned long long stride0) {
    u32x4 g0;
    g0[0] = 1u;                                   // count=1, user descriptor
    g0[1] = lds_off;                              // LDS byte address
    g0[2] = (unsigned int)gaddr;                  // global addr [31:0]
    g0[3] = (unsigned int)((gaddr >> 32) & 0x1FFFFFFu) | (2u << 30); // [56:32] | type=2
    i32x8 g1;
    g1[0] = (int)(1u << 16);                      // workgroup_mask=0, data_size=1 (2B)
    g1[1] = (int)((tile_d0 & 0xFFFFu) << 16);     // tensor_dim0[15:0] at bits 63:48
    g1[2] = (int)(((tile_d0 >> 16) & 0xFFFFu) | ((tile_d1 & 0xFFFFu) << 16)); // dim0 hi | dim1 lo
    g1[3] = (int)(((tile_d1 >> 16) & 0xFFFFu) | ((tile_d0 & 0xFFFFu) << 16)); // dim1 hi | tile_dim0
    g1[4] = (int)(tile_d1 & 0xFFFFu);             // tile_dim1 | tile_dim2=0
    g1[5] = (int)(stride0 & 0xFFFFFFFFull);       // tensor_dim0_stride lo32
    g1[6] = (int)((stride0 >> 32) & 0xFFFFull);   // stride hi16 | dim1_stride lo16 = 0
    g1[7] = 0;
    i32x4 z4 = {};
#if defined(__clang_major__) && (__clang_major__ >= 23)
    i32x8 z8 = {};
    __builtin_amdgcn_tensor_load_to_lds(g0, g1, z4, z4, z8, 0);
#else
    __builtin_amdgcn_tensor_load_to_lds(g0, g1, z4, z4, 0);
#endif
}

// ---------------------------------------------------------------------------
// Weight transpose + fp32 -> bf16 convert:  w[K,N] f32  ->  wt[N,K] bf16
// ---------------------------------------------------------------------------
__global__ __launch_bounds__(256) void wt_kernel(const float* __restrict__ w,
                                                 bf16* __restrict__ wt,
                                                 int K, int N) {
    size_t i = (size_t)blockIdx.x * 256 + threadIdx.x;
    size_t total = (size_t)K * N;
    if (i >= total) return;
    int n = (int)(i % N);
    int k = (int)(i / N);
    wt[(size_t)n * K + k] = (bf16)w[i];
}

// ---------------------------------------------------------------------------
// LayerNorm (row of 768) -> bf16
// ---------------------------------------------------------------------------
__global__ __launch_bounds__(256) void ln_kernel(const float* __restrict__ x,
                                                 const float* __restrict__ g,
                                                 const float* __restrict__ b,
                                                 bf16* __restrict__ out) {
    const int C = 768;
    int row = blockIdx.x;
    int t = threadIdx.x;
    const float* xr = x + (size_t)row * C;
    float v0 = xr[t], v1 = xr[t + 256], v2 = xr[t + 512];
    __shared__ float s1[256], s2[256];
    s1[t] = v0 + v1 + v2;
    s2[t] = v0 * v0 + v1 * v1 + v2 * v2;
    __syncthreads();
    for (int s = 128; s > 0; s >>= 1) {
        if (t < s) { s1[t] += s1[t + s]; s2[t] += s2[t + s]; }
        __syncthreads();
    }
    float mu = s1[0] * (1.0f / C);
    float var = s2[0] * (1.0f / C) - mu * mu;
    float rstd = rsqrtf(var + 1e-5f);
    bf16* orow = out + (size_t)row * C;
    orow[t]       = (bf16)((v0 - mu) * rstd * g[t]       + b[t]);
    orow[t + 256] = (bf16)((v1 - mu) * rstd * g[t + 256] + b[t + 256]);
    orow[t + 512] = (bf16)((v2 - mu) * rstd * g[t + 512] + b[t + 512]);
}

// ---------------------------------------------------------------------------
// bf16 WMMA GEMM:  out[M,N] = A[M,K] @ WT[N,K]^T  (+bias, +gelu, +residual)
// Block = 256 threads = 8 waves; wave tile 32(M) x 64(N) (2 A-frags x 4 B-frags
// -> 8 WMMA per 12 b128 loads); block tile 128 x 128.
// Requires M % 128 == 0, N % 128 == 0, K % 32 == 0.
// ---------------------------------------------------------------------------
__global__ __launch_bounds__(256) void gemm_kernel(const bf16* __restrict__ A,
                                                   const bf16* __restrict__ WT,
                                                   int M, int N, int K,
                                                   const float* __restrict__ bias,
                                                   const float* __restrict__ residual,
                                                   float* __restrict__ outF,
                                                   bf16* __restrict__ outB,
                                                   int fuse_gelu) {
    const int lane = threadIdx.x & 31;
    const int l15  = lane & 15;
    const int half = lane >> 4;
    const int wave = threadIdx.x >> 5;
    const int wm = wave & 3;      // 4 waves along M (32 rows each)
    const int wn = wave >> 2;     // 2 waves along N (64 cols each)
    const int row0 = blockIdx.y * 128 + wm * 32;
    const int col0 = blockIdx.x * 128 + wn * 64;

    v8f c[2][4] = {};
    const bf16* arow0 = A + (size_t)(row0 + l15) * K;
    const bf16* arow1 = arow0 + (size_t)16 * K;
    const bf16* bc[4];
#pragma unroll
    for (int t = 0; t < 4; t++)
        bc[t] = WT + (size_t)(col0 + t * 16 + l15) * K;

    for (int kc = 0; kc < K; kc += 32) {
        v16bf a0 = load_afrag(arow0 + kc, half);
        v16bf a1 = load_afrag(arow1 + kc, half);
#pragma unroll
        for (int t = 0; t < 4; t++) {
            v16bf bb = load_bfrag(bc[t] + kc, half);
            c[0][t] = wmma_bf16(a0, bb, c[0][t]);
            c[1][t] = wmma_bf16(a1, bb, c[1][t]);
        }
    }

#pragma unroll
    for (int r = 0; r < 2; r++) {
#pragma unroll
        for (int t = 0; t < 4; t++) {
            int col = col0 + t * 16 + l15;
            float bv = bias ? bias[col] : 0.0f;
#pragma unroll
            for (int j = 0; j < 8; j++) {
                int row = row0 + r * 16 + j + 8 * half;
                float v = c[r][t][j] + bv;
                if (fuse_gelu)
                    v = 0.5f * v * (1.0f + erff(v * 0.70710678118654752f));
                if (residual)
                    v += residual[(size_t)row * N + col];
                if (outF)
                    outF[(size_t)row * N + col] = v;
                else
                    outB[(size_t)row * N + col] = (bf16)v;
            }
        }
    }
}

// ---------------------------------------------------------------------------
// Extract + transpose V:  vt[bh][d][n] = qkv[(b*1024+n)*2304 + 1536 + h*64 + d]
// ---------------------------------------------------------------------------
__global__ __launch_bounds__(256) void vtrans_kernel(const bf16* __restrict__ qkv,
                                                     bf16* __restrict__ vt) {
    size_t i = (size_t)blockIdx.x * 256 + threadIdx.x;   // total = 96*64*1024
    if (i >= (size_t)96 * 64 * 1024) return;
    int n  = (int)(i & 1023);
    size_t r = i >> 10;
    int d  = (int)(r & 63);
    int bh = (int)(r >> 6);
    int b = bh / 12, h = bh % 12;
    vt[i] = qkv[((size_t)(b * 1024 + n)) * 2304 + 1536 + h * 64 + d];
}

// ---------------------------------------------------------------------------
// Flash attention: block = 4 waves (128 threads) sharing one (b,h), each wave
// owns a 16-query tile. Per 32-key step, wave 0 DMAs the K tile (32x64) and
// transposed V tile (64x32) into LDS via the Tensor Data Mover; all waves
// build WMMA fragments from LDS.
// ---------------------------------------------------------------------------
__global__ __launch_bounds__(128) void attn_kernel(const bf16* __restrict__ qkv,
                                                   const int* __restrict__ mask,
                                                   const bf16* __restrict__ vt,
                                                   bf16* __restrict__ o_out) {
    const int Ntok = 1024, C3 = 2304, H = 12, D = 64;
    const float scale = 0.125f;            // 1/sqrt(64)
    const int lane = threadIdx.x & 31;
    const int l15  = lane & 15;
    const int half = lane >> 4;
    const int wave = threadIdx.x >> 5;     // 0..3
    const int qt = blockIdx.x * 64 + wave * 16;
    const int bh = blockIdx.y;
    const int b = bh / H, h = bh % H;

    __shared__ __align__(16) bf16 Ktile[32 * 64];   // [key][dim]
    __shared__ __align__(16) bf16 Vtile[64 * 32];   // [dim][key]
    __shared__ __align__(16) bf16 P[4][16 * 32];    // per-wave probs

    // Q fragments (dims 0..31, 32..63)
    const bf16* qbase = qkv + ((size_t)(b * Ntok + qt + l15)) * C3 + h * D;
    v16bf a0 = load_afrag(qbase + 0,  half);
    v16bf a1 = load_afrag(qbase + 32, half);

    const bf16* kglob = qkv + (size_t)b * Ntok * C3 + 768 + h * D; // + key*C3
    const bf16* vglob = vt + (size_t)bh * D * Ntok;                // + d*1024 + key
    const int* mrow = mask + b * Ntok;

    float m[8], l[8];
#pragma unroll
    for (int j = 0; j < 8; j++) { m[j] = -3.0e38f; l[j] = 0.0f; }
    v8f o0 = {}, o1 = {}, o2 = {}, o3 = {};

    const unsigned int koff = (unsigned int)(size_t)(&Ktile[0]);
    const unsigned int voff = (unsigned int)(size_t)(&Vtile[0]);

    for (int k0 = 0; k0 < Ntok; k0 += 32) {
        __syncthreads();   // all waves done reading previous K/V tiles
        if (threadIdx.x < 32) {
            // K tile: 32 rows (keys) x 64 elems, row stride 2304 elems
            tdm_load_2d_bf16(koff,
                             (unsigned long long)(size_t)(kglob + (size_t)k0 * C3),
                             64u, 32u, (unsigned long long)C3);
            // V tile: 64 rows (dims) x 32 elems (keys), row stride 1024 elems
            tdm_load_2d_bf16(voff,
                             (unsigned long long)(size_t)(vglob + k0),
                             32u, 64u, (unsigned long long)Ntok);
            __builtin_amdgcn_s_wait_tensorcnt(0);
        }
        __syncthreads();   // tiles visible to all waves

        // --- scores for two 16-key sub-tiles from LDS K ----------------
        v8f s[2];
#pragma unroll
        for (int sub = 0; sub < 2; sub++) {
            const bf16* kp = &Ktile[(sub * 16 + l15) * 64];
            v8f acc = {};
            acc = wmma_bf16(a0, load_bfrag(kp + 0,  half), acc);
            acc = wmma_bf16(a1, load_bfrag(kp + 32, half), acc);
            int km = mrow[k0 + sub * 16 + l15];
#pragma unroll
            for (int j = 0; j < 8; j++)
                s[sub][j] = km ? acc[j] * scale : -1.0e9f;
        }
        // --- online softmax (row = j + 8*half per half-wave) -----------
#pragma unroll
        for (int j = 0; j < 8; j++) {
            float mx = halfmax(fmaxf(s[0][j], s[1][j]));
            float mn = fmaxf(m[j], mx);
            float alpha = __expf(m[j] - mn);
            s[0][j] = __expf(s[0][j] - mn);
            s[1][j] = __expf(s[1][j] - mn);
            float rs = halfsum(s[0][j] + s[1][j]);
            l[j] = l[j] * alpha + rs;
            m[j] = mn;
            o0[j] *= alpha; o1[j] *= alpha; o2[j] *= alpha; o3[j] *= alpha;
            P[wave][(j + 8 * half) * 32 + l15]      = (bf16)s[0][j];
            P[wave][(j + 8 * half) * 32 + 16 + l15] = (bf16)s[1][j];
        }
        __syncthreads();   // P store -> P load fence (and keeps waves together)

        v8bf plo = *reinterpret_cast<const v8bf*>(&P[wave][l15 * 32 + 8 * half]);
        v8bf phi = *reinterpret_cast<const v8bf*>(&P[wave][l15 * 32 + 16 + 8 * half]);
        v16bf pa = cat8(plo, phi);
        // --- O += P @ V from LDS V tile --------------------------------
        o0 = wmma_bf16(pa, load_bfrag(&Vtile[( 0 + l15) * 32], half), o0);
        o1 = wmma_bf16(pa, load_bfrag(&Vtile[(16 + l15) * 32], half), o1);
        o2 = wmma_bf16(pa, load_bfrag(&Vtile[(32 + l15) * 32], half), o2);
        o3 = wmma_bf16(pa, load_bfrag(&Vtile[(48 + l15) * 32], half), o3);
    }

    // --- epilogue: normalize and store bf16 ---------------------------
    v8f oacc[4] = {o0, o1, o2, o3};
#pragma unroll
    for (int j = 0; j < 8; j++) {
        int row = qt + j + 8 * half;
        float inv = 1.0f / l[j];
        bf16* orow = o_out + ((size_t)(b * Ntok + row)) * 768 + h * D;
#pragma unroll
        for (int t = 0; t < 4; t++)
            orow[t * 16 + l15] = (bf16)(oacc[t][j] * inv);
    }
}

// ---------------------------------------------------------------------------
// Host-side orchestration
// ---------------------------------------------------------------------------
extern "C" void kernel_launch(void* const* d_in, const int* in_sizes, int n_in,
                              void* d_out, int out_size, void* d_ws, size_t ws_size,
                              hipStream_t stream) {
    const float* x      = (const float*)d_in[0];
    const int*   mask   = (const int*)  d_in[1];
    const float* g1     = (const float*)d_in[2];
    const float* b1     = (const float*)d_in[3];
    const float* w_qkv  = (const float*)d_in[4];
    const float* w_proj = (const float*)d_in[5];
    const float* b_proj = (const float*)d_in[6];
    const float* g2     = (const float*)d_in[7];
    const float* b2     = (const float*)d_in[8];
    const float* w_fc1  = (const float*)d_in[9];
    const float* b_fc1  = (const float*)d_in[10];
    const float* w_fc2  = (const float*)d_in[11];
    const float* b_fc2  = (const float*)d_in[12];
    float* out = (float*)d_out;

    const int M = 8192, C = 768, C3 = 2304;

    char* ws = (char*)d_ws;
    size_t off = 0;
    auto alloc = [&](size_t bytes) -> void* {
        void* p = ws + off;
        off = (off + bytes + 255) & ~(size_t)255;
        return p;
    };
    bf16* wqkvT  = (bf16*)alloc((size_t)C3 * C * sizeof(bf16));
    bf16* wprojT = (bf16*)alloc((size_t)C * C * sizeof(bf16));
    bf16* wfc1T  = (bf16*)alloc((size_t)C * C * sizeof(bf16));
    bf16* wfc2T  = (bf16*)alloc((size_t)C * C * sizeof(bf16));
    bf16* h1     = (bf16*)alloc((size_t)M * C * sizeof(bf16));
    bf16* qkvb   = (bf16*)alloc((size_t)M * C3 * sizeof(bf16));
    bf16* vtb    = (bf16*)alloc((size_t)96 * 64 * 1024 * sizeof(bf16));
    bf16* attno  = (bf16*)alloc((size_t)M * C * sizeof(bf16));
    float* x2    = (float*)alloc((size_t)M * C * sizeof(float));
    bf16* h2     = (bf16*)alloc((size_t)M * C * sizeof(bf16));
    bf16* h3     = (bf16*)alloc((size_t)M * C * sizeof(bf16));

    // 1) weights -> transposed bf16
    wt_kernel<<<((size_t)C * C3 + 255) / 256, 256, 0, stream>>>(w_qkv, wqkvT, C, C3);
    wt_kernel<<<((size_t)C * C + 255) / 256, 256, 0, stream>>>(w_proj, wprojT, C, C);
    wt_kernel<<<((size_t)C * C + 255) / 256, 256, 0, stream>>>(w_fc1, wfc1T, C, C);
    wt_kernel<<<((size_t)C * C + 255) / 256, 256, 0, stream>>>(w_fc2, wfc2T, C, C);

    // 2) LN1 -> h1 (bf16)
    ln_kernel<<<M, 256, 0, stream>>>(x, g1, b1, h1);

    // 3) QKV GEMM: qkvb = h1 @ w_qkv   (bf16 out, no bias)
    gemm_kernel<<<dim3(C3 / 128, M / 128), 256, 0, stream>>>(
        h1, wqkvT, M, C3, C, nullptr, nullptr, nullptr, qkvb, 0);

    // 4) V transpose -> vt[bh][d][n]
    vtrans_kernel<<<((size_t)96 * 64 * 1024 + 255) / 256, 256, 0, stream>>>(qkvb, vtb);

    // 5) flash attention -> attno (bf16, [8192,768])
    attn_kernel<<<dim3(1024 / 64, 96), 128, 0, stream>>>(qkvb, mask, vtb, attno);

    // 6) proj GEMM + bias + residual(x) -> x2 (f32)
    gemm_kernel<<<dim3(C / 128, M / 128), 256, 0, stream>>>(
        attno, wprojT, M, C, C, b_proj, x, x2, nullptr, 0);

    // 7) LN2 -> h2 (bf16)
    ln_kernel<<<M, 256, 0, stream>>>(x2, g2, b2, h2);

    // 8) FC1 GEMM + bias + exact GELU -> h3 (bf16)
    gemm_kernel<<<dim3(C / 128, M / 128), 256, 0, stream>>>(
        h2, wfc1T, M, C, C, b_fc1, nullptr, nullptr, h3, 1);

    // 9) FC2 GEMM + bias + residual(x2) -> d_out (f32)
    gemm_kernel<<<dim3(C / 128, M / 128), 256, 0, stream>>>(
        h3, wfc2T, M, C, C, b_fc2, x2, out, nullptr, 0);
}
